// Model_14499809592071
// MI455X (gfx1250) — matrix operand order repaired and checked by reading, not compiled, not run on hardware
//
#include <hip/hip_runtime.h>

typedef __attribute__((ext_vector_type(16))) _Float16 v16h;
typedef __attribute__((ext_vector_type(8)))  float    v8f;

#define DD      32
#define HS      36                 // padded f32 row stride (bank-conflict free)
#define PMAXC   118
#define NMAXC   236
#define NSTATES (PMAXC + NMAXC)   // 354
#define WAVES   8
#define BLOCK   256
#define TPW     8                  // 16-row tiles per wave

__device__ __forceinline__ float silu_f(float v) {
    return v * __builtin_amdgcn_rcpf(1.0f + __expf(-v));
}

// Compiler-only reorder fence: per-wave LDS staging relies on CDNA5's
// in-order LDS pipeline for cross-lane visibility inside one wave.
__device__ __forceinline__ void wave_lds_order() {
    asm volatile("" ::: "memory");
}

// ---------------------------------------------------------------------------
// Phase 1: sequential SiLU chains -> f16 state tables in workspace.
// ---------------------------------------------------------------------------
__global__ void chain_kernel(const float* __restrict__ emb,
                             const float* __restrict__ Wp, const float* __restrict__ bp,
                             const float* __restrict__ Wn, const float* __restrict__ bn,
                             _Float16* __restrict__ states)
{
    __shared__ float sv[DD];
    const int t = threadIdx.x;            // 32 threads
    float wrow[DD];

    #pragma unroll
    for (int k = 0; k < DD; ++k) wrow[k] = Wp[t * DD + k];
    float p = emb[t];                     // embeddings[0]
    const float bpt = bp[t];
    for (int step = 0; step < PMAXC; ++step) {
        sv[t] = silu_f(p);
        __syncthreads();
        float acc = bpt;
        #pragma unroll
        for (int k = 0; k < DD; ++k) acc += sv[k] * wrow[k];
        __syncthreads();
        p = acc;
        states[step * DD + t] = (_Float16)p;
    }

    #pragma unroll
    for (int k = 0; k < DD; ++k) wrow[k] = Wn[t * DD + k];
    p = emb[DD + t];                      // embeddings[1]
    const float bnt = bn[t];
    for (int step = 0; step < NMAXC; ++step) {
        sv[t] = silu_f(p);
        __syncthreads();
        float acc = bnt;
        #pragma unroll
        for (int k = 0; k < DD; ++k) acc += sv[k] * wrow[k];
        __syncthreads();
        p = acc;
        states[(PMAXC + step) * DD + t] = (_Float16)p;
    }
}

// ---------------------------------------------------------------------------
// Phase 2: gather + GEMM1 on WMMA; LN + (W2,ln,Wr) all folded into a
// rank-2 reduction:  out[o] = inv*(P[o] - mu*Sw[o]) + beff[o].
// ---------------------------------------------------------------------------
__global__ __launch_bounds__(BLOCK) void fused_mlp_kernel(
    const int*      __restrict__ x,
    const _Float16* __restrict__ statesG,
    const float* __restrict__ W1, const float* __restrict__ b1,
    const float* __restrict__ lng, const float* __restrict__ lnb,
    const float* __restrict__ W2, const float* __restrict__ b2,
    const float* __restrict__ Wr, const float* __restrict__ br,
    float* __restrict__ out)
{
    __shared__ _Float16 sStates[NSTATES * DD];        // 22656 B f16 state tables
    __shared__ float    sWeff[2 * DD];                // Wr @ (g*W2)
    __shared__ float    sC2[DD];                      // b2 + W2@lnb
    __shared__ float    sConst[4];                    // beff0, beff1, Sw0, Sw1
    __shared__ float    sH[WAVES][16 * HS];           // per-wave f32 staging

    const int tid = threadIdx.x;

    // cooperative copy of the f16 state tables (128-bit chunks)
    {
        const uint4* src = (const uint4*)statesG;
        uint4*       dst = (uint4*)sStates;
        for (int i = tid; i < NSTATES * DD * 2 / 16; i += BLOCK) dst[i] = src[i];
    }
    // Weff[o][k] = sum_n Wr[o][n]*W2[n][k] * g[k]
    if (tid < 2 * DD) {
        const int o = tid >> 5, k = tid & 31;
        float acc = 0.f;
        #pragma unroll
        for (int n = 0; n < DD; ++n) acc += Wr[o * DD + n] * W2[n * DD + k];
        sWeff[o * DD + k] = acc * lng[k];
    }
    // c2[n] = b2[n] + sum_k W2[n][k]*lnb[k]
    if (tid >= 64 && tid < 64 + DD) {
        const int n = tid - 64;
        float acc = b2[n];
        #pragma unroll
        for (int k = 0; k < DD; ++k) acc += W2[n * DD + k] * lnb[k];
        sC2[n] = acc;
    }
    __syncthreads();
    // beff[o] = br[o] + Wr[o]·c2 ; Sw[o] = sum_k Weff[o][k]
    if (tid < 2) {
        float acc = br[tid], sw = 0.f;
        #pragma unroll
        for (int n = 0; n < DD; ++n) {
            acc += Wr[tid * DD + n] * sC2[n];
            sw  += sWeff[tid * DD + n];
        }
        sConst[tid]     = acc;
        sConst[2 + tid] = sw;
    }
    __syncthreads();

    const int wave = tid >> 5;
    const int lane = tid & 31;
    const int l16  = lane & 15;
    const int hi   = lane >> 4;

    const long waveTile0 = ((long)blockIdx.x * WAVES + wave) * TPW;

    // ---- GEMM1 B fragments (K x N=16), kept in VGPRs for all tiles ----
    // element e of lane -> K_local = (e&7) + ((e>>3)<<4) + 8*hi, N = l16 (+16 for second half)
    v16h B1f[2][2];   // [K-chunk (pf/nf)][N-half]
    #pragma unroll
    for (int c = 0; c < 2; ++c)
        #pragma unroll
        for (int h = 0; h < 2; ++h) {
            #pragma unroll
            for (int e = 0; e < 16; ++e) {
                const int k = c * 32 + (e & 7) + ((e >> 3) << 4) + (hi << 3);
                const int n = h * 16 + l16;
                B1f[c][h][e] = (_Float16)W1[n * (2 * DD) + k];
            }
        }

    // per-lane constants / weight slices
    const float b1lo = b1[l16], b1hi = b1[16 + l16];
    float w0[16], w1[16];
    #pragma unroll
    for (int n0 = 0; n0 < 16; ++n0) {
        w0[n0] = sWeff[hi * 16 + n0];
        w1[n0] = sWeff[DD + hi * 16 + n0];
    }
    const float beff0 = sConst[0], beff1 = sConst[1];
    const float Sw0   = sConst[2], Sw1   = sConst[3];

    float* H = sH[wave];

    // software-pipelined x indices: load tile t+1 while processing tile t
    int2 xcur = ((const int2*)x)[waveTile0 * 16 + l16];

    for (int t = 0; t < TPW; ++t) {
        const int2 xnext = (t + 1 < TPW)
                         ? ((const int2*)x)[(waveTile0 + t + 1) * 16 + l16]
                         : xcur;

        const long row0 = (waveTile0 + t) * 16;
        const _Float16* pRow = &sStates[(xcur.x - 1) * DD];
        const _Float16* nRow = &sStates[(PMAXC + xcur.y - 1) * DD];

        // ---- A fragments (16x32 f16): K(e,lane) = (e&7) + ((e>>3)<<4) + (hi<<3)
        v16h A0, A1;
        #pragma unroll
        for (int e = 0; e < 16; e += 2) {
            const int K = (e & 7) + ((e >> 3) << 4) + (hi << 3);
            A0[e] = pRow[K];  A0[e + 1] = pRow[K + 1];
            A1[e] = nRow[K];  A1[e + 1] = nRow[K + 1];
        }

        // ---- GEMM1: h1(16x32) = [pf|nf](16x64) @ W1^T, f32 accumulate
        v8f C0 = {}, C1 = {};
        C0 = __builtin_amdgcn_wmma_f32_16x16x32_f16(false, A0, false, B1f[0][0], (short)0, C0, false, false);
        C0 = __builtin_amdgcn_wmma_f32_16x16x32_f16(false, A1, false, B1f[1][0], (short)0, C0, false, false);
        C1 = __builtin_amdgcn_wmma_f32_16x16x32_f16(false, A0, false, B1f[0][1], (short)0, C1, false, false);
        C1 = __builtin_amdgcn_wmma_f32_16x16x32_f16(false, A1, false, B1f[1][1], (short)0, C1, false, false);

        // ---- epilogue: +b1, SiLU -> f32 staging (padded rows, conflict-free)
        #pragma unroll
        for (int r = 0; r < 8; ++r) {
            const int row = r + 8 * hi;          // C/D layout: vgpr r, lane half hi
            H[row * HS + l16]      = silu_f(C0[r] + b1lo);
            H[row * HS + 16 + l16] = silu_f(C1[r] + b1hi);
        }
        wave_lds_order();

        // ---- fused LN + rank-2 head: 2 lanes per row (row=l16, cols hi*16..)
        {
            float s = 0.f, ss = 0.f, p0 = 0.f, p1 = 0.f;
            #pragma unroll
            for (int n0 = 0; n0 < 16; ++n0) {
                const float h = H[l16 * HS + hi * 16 + n0];
                s  += h;
                ss += h * h;
                p0 += h * w0[n0];
                p1 += h * w1[n0];
            }
            s  += __shfl_xor(s,  16, 32);
            ss += __shfl_xor(ss, 16, 32);
            p0 += __shfl_xor(p0, 16, 32);
            p1 += __shfl_xor(p1, 16, 32);

            const float mu  = s * (1.0f / DD);
            float       var = ss * (1.0f / DD) - mu * mu;
            var = fmaxf(var, 0.0f);
            const float inv = __builtin_amdgcn_rsqf(var + 1e-5f);

            if (hi == 0) {
                const float o0 = inv * (p0 - mu * Sw0) + beff0;
                const float o1 = inv * (p1 - mu * Sw1) + beff1;
                ((float2*)out)[row0 + l16] = make_float2(o0, o1);
            }
        }
        wave_lds_order();

        xcur = xnext;
    }
}

// ---------------------------------------------------------------------------
extern "C" void kernel_launch(void* const* d_in, const int* in_sizes, int n_in,
                              void* d_out, int out_size, void* d_ws, size_t ws_size,
                              hipStream_t stream)
{
    const int*   x   = (const int*)  d_in[0];
    const float* emb = (const float*)d_in[1];
    const float* Wp  = (const float*)d_in[2];
    const float* bp  = (const float*)d_in[3];
    const float* Wn  = (const float*)d_in[4];
    const float* bn  = (const float*)d_in[5];
    const float* W1  = (const float*)d_in[6];
    const float* b1  = (const float*)d_in[7];
    const float* lng = (const float*)d_in[8];
    const float* lnb = (const float*)d_in[9];
    const float* W2  = (const float*)d_in[10];
    const float* b2  = (const float*)d_in[11];
    const float* Wr  = (const float*)d_in[12];
    const float* br  = (const float*)d_in[13];
    float*    out    = (float*)d_out;
    _Float16* states = (_Float16*)d_ws;   // 354*32 f16 = 22656 B

    chain_kernel<<<1, 32, 0, stream>>>(emb, Wp, bp, Wn, bn, states);

    const long B = (long)in_sizes[0] / 2;                 // rows (x is B x 2)
    const long rowsPerBlock = (long)WAVES * TPW * 16;     // 1024
    const int  blocks = (int)((B + rowsPerBlock - 1) / rowsPerBlock);

    fused_mlp_kernel<<<blocks, BLOCK, 0, stream>>>(x, states, W1, b1, lng, lnb,
                                                   W2, b2, Wr, br, out);
}
